// MOE_IFM_Clip_63531156242762
// MI455X (gfx1250) — compile-verified, matrix operand order checked by reading, not run on hardware
//
#include <hip/hip_runtime.h>
#include <hip/hip_bf16.h>
#include <math.h>

// ---------------------------------------------------------------------------
// MoE-IFM for MI455X (gfx1250): conv3x3 as 9 tap-GEMMs via
// V_WMMA_F32_16X16X32_BF16. All WMMA fragments are contiguous 32B LDS chunks
// (pixel-major activations, [tap][co][ci] weights) -> ds_load_b128 fragment
// builds. Top-2 sparse expert dispatch, register-resident exp-combine.
// ---------------------------------------------------------------------------

typedef __attribute__((ext_vector_type(16))) __bf16 v16bf;
typedef __attribute__((ext_vector_type(8)))  __bf16 v8bf;
typedef __attribute__((ext_vector_type(8)))  float  v8f;

#define B_    16
#define C_    32
#define E_    8
#define H_    128
#define W_    128
#define HW_   (H_ * W_)
#define TILE_ 16
#define REG_  20              // 16 + 2*2 halo
#define RS_   (REG_ * REG_)   // 400 region pixels
#define NGRP_ (RS_ / 16)      // 25 pixel groups
#define PLANE_ ((size_t)B_ * C_ * HW_)

#define EPS_F 2.220446049250313e-16f

// LDS layout (bf16 element offsets). Activations pixel-major: [p][c].
#define XS_OFF   0
#define PS_OFF   (XS_OFF + RS_ * C_)
#define AXS_OFF  (PS_OFF + RS_ * C_)
#define APS_OFF  (AXS_OFF + RS_ * C_)
#define WQ_OFF   (APS_OFF + RS_ * C_)        // [tap][co32][ci32]
#define WK_OFF   (WQ_OFF + 9 * C_ * C_)
#define WG1_OFF  (WK_OFF + 9 * C_ * C_)      // [tap][m4][ch64], rows 2,3 zero
#define WG2_OFF  (WG1_OFF + 9 * 4 * 64)
#define BF_END   (WG2_OFF + 9 * 4 * 64)
#define SMEM_BYTES (BF_END * 2 + 4 * 256 * 4)

__device__ __forceinline__ v8f wmma_bf16(v16bf a, v16bf b, v8f c) {
  return __builtin_amdgcn_wmma_f32_16x16x32_bf16(
      false, a, false, b, (short)0, c, false, false);
}
__device__ __forceinline__ float sigmoidf_(float z) {
  return __builtin_amdgcn_rcpf(1.0f + __expf(-z));   // v_rcp_f32 path
}
__device__ __forceinline__ int clampi(int v, int lo, int hi) {
  return v < lo ? lo : (v > hi ? hi : v);
}

// ---------------------------------------------------------------------------
// Kernel 1: pooled[b][c] = mean over H,W
// ---------------------------------------------------------------------------
__global__ void moe_pool_kernel(const float* __restrict__ x,
                                float* __restrict__ pooled) {
  __shared__ float red[256];
  const int bc = blockIdx.x;
  const float* base = x + (size_t)bc * HW_;
  float s = 0.f;
  for (int i = threadIdx.x; i < HW_; i += 256) s += base[i];
  red[threadIdx.x] = s;
  __syncthreads();
  for (int st = 128; st > 0; st >>= 1) {
    if (threadIdx.x < st) red[threadIdx.x] += red[threadIdx.x + st];
    __syncthreads();
  }
  if (threadIdx.x == 0) pooled[bc] = red[0] * (1.0f / (float)HW_);
}

// ---------------------------------------------------------------------------
// Kernel 2: Shazeer noisy top-2 gating, one thread per batch row
// ---------------------------------------------------------------------------
__global__ void moe_gate_kernel(const float* __restrict__ pooled,
                                const float* __restrict__ noise,
                                const float* __restrict__ w_gate,
                                const float* __restrict__ w_noise,
                                float* __restrict__ gates) {
  const int b = threadIdx.x;
  if (b >= B_) return;
  float logit[E_];
#pragma unroll
  for (int e = 0; e < E_; ++e) {
    float cl = 0.f, nv = 0.f;
#pragma unroll
    for (int c = 0; c < C_; ++c) {
      const float pv = pooled[b * C_ + c];
      cl += pv * w_gate[c * E_ + e];
      nv += pv * w_noise[c * E_ + e];
    }
    const float nstd = log1pf(__expf(nv)) + 1e-2f;   // softplus + NOISE_EPS
    logit[e] = cl + noise[b * E_ + e] * nstd;
  }
  int i1 = 0;
#pragma unroll
  for (int e = 1; e < E_; ++e) if (logit[e] > logit[i1]) i1 = e;
  int i2 = (i1 == 0) ? 1 : 0;
#pragma unroll
  for (int e = 0; e < E_; ++e)
    if (e != i1 && logit[e] > logit[i2]) i2 = e;
  const float e2  = __expf(logit[i2] - logit[i1]);
  const float inv = __builtin_amdgcn_rcpf(1.0f + e2);
#pragma unroll
  for (int e = 0; e < E_; ++e) gates[b * E_ + e] = 0.f;
  gates[b * E_ + i1] = inv;
  gates[b * E_ + i2] = e2 * inv;
}

// ---------------------------------------------------------------------------
// Kernel 3: per-(batch, 16x16 tile) IFM over active experts.
// Block = 256 threads = 8 wave32.
// ---------------------------------------------------------------------------
__global__ void __launch_bounds__(256)
moe_ifm_kernel(const float* __restrict__ x, const float* __restrict__ prior,
               const float* __restrict__ Wq, const float* __restrict__ bq,
               const float* __restrict__ Wk, const float* __restrict__ bk,
               const float* __restrict__ Wg1, const float* __restrict__ bg1,
               const float* __restrict__ Wg2, const float* __restrict__ bg2,
               const float* __restrict__ gates, float* __restrict__ out) {
  extern __shared__ char smem[];
  __bf16* Xs   = (__bf16*)smem;                 // [400][32] pixel-major
  __bf16* Ps   = Xs + PS_OFF;
  __bf16* AxS  = Xs + AXS_OFF;
  __bf16* ApS  = Xs + APS_OFF;
  __bf16* WqS  = Xs + WQ_OFF;                   // [tap][co][ci]
  __bf16* WkS  = Xs + WK_OFF;
  __bf16* Wg1S = Xs + WG1_OFF;                  // [tap][m4][ch64]
  __bf16* Wg2S = Xs + WG2_OFF;
  float* XG0 = (float*)(Xs + BF_END);
  float* XG1 = XG0 + 256;
  float* PG0 = XG1 + 256;
  float* PG1 = PG0 + 256;

  const int tid  = threadIdx.x;
  const int lane = tid & 31;
  const int wv   = tid >> 5;
  const int tY   = blockIdx.x >> 3;
  const int tX   = blockIdx.x & 7;
  const int b    = blockIdx.y;

  // ---- stage input tiles: 8 channels per b128 store --------------------
  for (int idx = tid; idx < RS_ * 4; idx += 256) {
    const int p  = idx >> 2;
    const int cg = (idx & 3) * 8;
    const int py = p / REG_, px = p % REG_;
    const int gy = tY * TILE_ - 2 + py, gx = tX * TILE_ - 2 + px;
    const bool in = (gy >= 0 && gy < H_ && gx >= 0 && gx < W_);
    v8bf xv, pv;
#pragma unroll
    for (int j = 0; j < 8; ++j) {
      float a = 0.f, q = 0.f;
      if (in) {
        const size_t o = ((size_t)(b * C_ + cg + j)) * HW_ + (size_t)gy * W_ + gx;
        a = x[o];
        q = prior[o];
      }
      xv[j] = (__bf16)a;
      pv[j] = (__bf16)q;
    }
    *(v8bf*)(Xs + p * C_ + cg) = xv;
    *(v8bf*)(Ps + p * C_ + cg) = pv;
  }

  float accX[32], accP[32];
#pragma unroll
  for (int i = 0; i < 32; ++i) { accX[i] = 0.f; accP[i] = 0.f; }

  __syncthreads();

  for (int e = 0; e < E_; ++e) {
    const float gv = gates[b * E_ + e];   // uniform across block
    if (gv == 0.f) continue;              // top-2 sparse: skip dead experts

    // ---- stage expert weights: [tap][co][ci], 8 ci per b128 store ------
    for (int idx = tid; idx < 9 * C_ * 4; idx += 256) {   // 1152 chunks
      const int tap = idx >> 7;
      const int rem = idx & 127;
      const int co  = rem >> 2;
      const int cg  = (rem & 3) * 8;
      v8bf wq, wk;
#pragma unroll
      for (int j = 0; j < 8; ++j) {
        const size_t g = (size_t)e * (C_ * C_ * 9) +
                         (size_t)(co * C_ + cg + j) * 9 + tap;
        wq[j] = (__bf16)Wq[g];
        wk[j] = (__bf16)Wk[g];
      }
      *(v8bf*)(WqS + tap * (C_ * C_) + co * C_ + cg) = wq;
      *(v8bf*)(WkS + tap * (C_ * C_) + co * C_ + cg) = wk;
    }
    for (int idx = tid; idx < 9 * 4 * 8; idx += 256) {    // 288 chunks
      const int tap = idx >> 5;
      const int rem = idx & 31;
      const int m   = rem >> 3;
      const int cg  = (rem & 7) * 8;
      v8bf w1, w2;
#pragma unroll
      for (int j = 0; j < 8; ++j) {
        const int ch = cg + j;
        float a = 0.f, q = 0.f;
        if (m < 2) {
          const size_t g = (size_t)e * 1152 + (size_t)(m * 64 + ch) * 9 + tap;
          a = Wg1[g];
          q = Wg2[g];
        }
        w1[j] = (__bf16)a;
        w2[j] = (__bf16)q;
      }
      *(v8bf*)(Wg1S + tap * 256 + m * 64 + cg) = w1;
      *(v8bf*)(Wg2S + tap * 256 + m * 64 + cg) = w2;
    }
    if (e + 1 < E_) {
      __builtin_prefetch(Wq + (size_t)(e + 1) * (C_ * C_ * 9), 0, 1);
    }
    __syncthreads();

    // ---- Phase A: q/k convs over 20x20 region, att, ax/ap --------------
    const int n  = lane & 15;            // WMMA column / A row
    const int kb = (lane >> 4) * 16;     // K half owned by this lane
    const int h8 = (lane >> 4) * 8;      // C/D row base: M = v + h8
    for (int g = wv; g < NGRP_; g += 8) {
      const int p  = g * 16 + n;
      const int py = p / REG_, px = p % REG_;
      v8f q0 = {}, q1 = {}, k0 = {}, k1 = {};
      for (int tap = 0; tap < 9; ++tap) {
        const int dy = tap / 3 - 1, dx = tap % 3 - 1;
        const int sy = clampi(py + dy, 0, REG_ - 1);
        const int sx = clampi(px + dx, 0, REG_ - 1);
        const int sp = sy * REG_ + sx;
        const v16bf bx = *(const v16bf*)(Xs + sp * C_ + kb);
        const v16bf bp = *(const v16bf*)(Ps + sp * C_ + kb);
        const __bf16* wbase = WqS + tap * (C_ * C_);
        const v16bf a0 = *(const v16bf*)(wbase + n * C_ + kb);
        const v16bf a1 = *(const v16bf*)(wbase + (16 + n) * C_ + kb);
        const __bf16* kbase = WkS + tap * (C_ * C_);
        const v16bf c0 = *(const v16bf*)(kbase + n * C_ + kb);
        const v16bf c1 = *(const v16bf*)(kbase + (16 + n) * C_ + kb);
        q0 = wmma_bf16(a0, bx, q0);
        q1 = wmma_bf16(a1, bx, q1);
        k0 = wmma_bf16(c0, bp, k0);
        k1 = wmma_bf16(c1, bp, k1);
      }
      // att = sigmoid((q+bq)*(k+bk)); ax/ap written as b128 chunks
      const v8bf xlo = *(const v8bf*)(Xs + p * C_ + h8);
      const v8bf xhi = *(const v8bf*)(Xs + p * C_ + 16 + h8);
      const v8bf plo = *(const v8bf*)(Ps + p * C_ + h8);
      const v8bf phi = *(const v8bf*)(Ps + p * C_ + 16 + h8);
      v8bf axlo, axhi, aplo, aphi;
#pragma unroll
      for (int v = 0; v < 8; ++v) {
        const int co = h8 + v;
        const float at0 = sigmoidf_((q0[v] + bq[e * C_ + co]) *
                                    (k0[v] + bk[e * C_ + co]));
        const float at1 = sigmoidf_((q1[v] + bq[e * C_ + co + 16]) *
                                    (k1[v] + bk[e * C_ + co + 16]));
        axlo[v] = (__bf16)((float)xlo[v] * at0);
        aplo[v] = (__bf16)((float)plo[v] * at0);
        axhi[v] = (__bf16)((float)xhi[v] * at1);
        aphi[v] = (__bf16)((float)phi[v] * at1);
      }
      *(v8bf*)(AxS + p * C_ + h8)      = axlo;
      *(v8bf*)(AxS + p * C_ + 16 + h8) = axhi;
      *(v8bf*)(ApS + p * C_ + h8)      = aplo;
      *(v8bf*)(ApS + p * C_ + 16 + h8) = aphi;
    }
    __syncthreads();

    // ---- Phase B: 64->2 gating convs over 16x16 interior ----------------
    {
      const int mc = (n < 3) ? n : 3;   // rows 2,3 of WgS are zero padding
      for (int g2 = wv; g2 < 16; g2 += 8) {
        const int ip = g2 * 16 + n;
        const int iy = ip >> 4, ix = ip & 15;
        const int p  = (iy + 2) * REG_ + (ix + 2);
        v8f xg = {}, pg = {};
        for (int tap = 0; tap < 9; ++tap) {
          const int dy = tap / 3 - 1, dx = tap % 3 - 1;
          const int sp = p + dy * REG_ + dx;        // always in region
#pragma unroll
          for (int kt = 0; kt < 2; ++kt) {          // concat halves: x | ax
            const __bf16* bsrc = (kt == 0) ? Xs  : AxS;
            const __bf16* qsrc = (kt == 0) ? Ps  : ApS;
            const v16bf bb  = *(const v16bf*)(bsrc + sp * C_ + kb);
            const v16bf bqf = *(const v16bf*)(qsrc + sp * C_ + kb);
            const int wi = tap * 256 + mc * 64 + kt * 32 + kb;
            const v16bf a1f = *(const v16bf*)(Wg1S + wi);
            const v16bf a2f = *(const v16bf*)(Wg2S + wi);
            xg = wmma_bf16(a1f, bb,  xg);
            pg = wmma_bf16(a2f, bqf, pg);
          }
        }
        if (lane < 16) {   // rows 0,1 -> acc elements 0,1 of lanes 0..15
          XG0[ip] = xg[0] + bg1[e * 2 + 0];
          XG1[ip] = xg[1] + bg1[e * 2 + 1];
          PG0[ip] = pg[0] + bg2[e * 2 + 0];
          PG1[ip] = pg[1] + bg2[e * 2 + 1];
        }
      }
    }
    __syncthreads();

    // ---- Phase C: gate-weighted exp-combine (one pixel per thread) -----
    {
      const int ip = tid;
      const int iy = ip >> 4, ix = ip & 15;
      const int p  = (iy + 2) * REG_ + (ix + 2);
      const float xg0 = XG0[ip], xg1 = XG1[ip];
      const float pg0 = PG0[ip], pg1 = PG1[ip];
#pragma unroll
      for (int k = 0; k < 4; ++k) {
        const v8bf xc  = *(const v8bf*)(Xs  + p * C_ + k * 8);
        const v8bf pc  = *(const v8bf*)(Ps  + p * C_ + k * 8);
        const v8bf axc = *(const v8bf*)(AxS + p * C_ + k * 8);
        const v8bf apc = *(const v8bf*)(ApS + p * C_ + k * 8);
#pragma unroll
        for (int v = 0; v < 8; ++v) {
          const int c = k * 8 + v;
          const float xo = (float)xc[v] * xg0 + (float)axc[v] * xg1;
          const float po = (float)pc[v] * pg0 + (float)apc[v] * pg1;
          accX[c] += gv * __expf(xo);
          accP[c] += gv * __expf(po);
        }
      }
    }
    __syncthreads();   // protect AxS/WqS/XG* before next expert
  }

  // ---- final: log(c==0 ? eps : c) -> out[2,B,C,H,W] ----------------------
  {
    const int ip = tid;
    const int iy = ip >> 4, ix = ip & 15;
    const int gy = tY * TILE_ + iy, gx = tX * TILE_ + ix;
    const size_t base = (size_t)b * C_ * HW_ + (size_t)gy * W_ + gx;
#pragma unroll
    for (int c = 0; c < 32; ++c) {
      const size_t o = base + (size_t)c * HW_;
      const float cx = accX[c], cp = accP[c];
      out[o]          = __logf(cx == 0.f ? EPS_F : cx);
      out[PLANE_ + o] = __logf(cp == 0.f ? EPS_F : cp);
    }
  }
}

// ---------------------------------------------------------------------------
extern "C" void kernel_launch(void* const* d_in, const int* in_sizes, int n_in,
                              void* d_out, int out_size, void* d_ws, size_t ws_size,
                              hipStream_t stream) {
  (void)in_sizes; (void)n_in; (void)out_size; (void)ws_size;
  const float* x       = (const float*)d_in[0];
  const float* prior   = (const float*)d_in[1];
  const float* noise   = (const float*)d_in[2];
  const float* w_gate  = (const float*)d_in[3];
  const float* w_noise = (const float*)d_in[4];
  const float* Wq      = (const float*)d_in[5];
  const float* bq      = (const float*)d_in[6];
  const float* Wk      = (const float*)d_in[7];
  const float* bk      = (const float*)d_in[8];
  const float* Wg1     = (const float*)d_in[9];
  const float* bg1     = (const float*)d_in[10];
  const float* Wg2     = (const float*)d_in[11];
  const float* bg2     = (const float*)d_in[12];
  float* out = (float*)d_out;

  float* pooled = (float*)d_ws;         // [B*C]
  float* gates  = pooled + B_ * C_;     // [B*E]

  moe_pool_kernel<<<B_ * C_, 256, 0, stream>>>(x, pooled);
  moe_gate_kernel<<<1, 32, 0, stream>>>(pooled, noise, w_gate, w_noise, gates);
  moe_ifm_kernel<<<dim3((H_ / TILE_) * (W_ / TILE_), B_), 256, SMEM_BYTES, stream>>>(
      x, prior, Wq, bq, Wk, bk, Wg1, bg1, Wg2, bg2, gates, out);
}
